// RPNHead_41377714929932
// MI455X (gfx1250) — compile-verified, matrix operand order checked by reading, not run on hardware
//
#include <hip/hip_runtime.h>
#include <hip/hip_bf16.h>

typedef __attribute__((ext_vector_type(16))) _Float16 v16h;
typedef __attribute__((ext_vector_type(8)))  _Float16 v8h;
typedef __attribute__((ext_vector_type(8)))  float    v8f;

#define NLVL 5
#define CCH 256
#define KTOT (9 * 256)
#define NCHUNK 72
#define TOTANCH 102300
#define TILES_PER_IMG 2132
#define PRE_NMS_K 1000
#define POST_NMS_K 300
#define NMS_TH 0.7f
#define MIN_SZ 0.1f
#define NEGV (-1.0f)
#define BBOX_CLIP_F 4.135166556742356f

__constant__ int   cHS[5]     = {160, 80, 40, 20, 10};
__constant__ float cSTRIDE[5] = {4.f, 8.f, 16.f, 32.f, 64.f};
__constant__ float cSIZE[5]   = {32.f, 64.f, 128.f, 256.f, 512.f};
__constant__ int   cLOFF[5]   = {0, 76800, 96000, 100800, 102000};

// Load a 16-half WMMA fragment from two contiguous 16B runs (K and K+16).
static __device__ inline v16h frag16(const _Float16* p) {
    v8h lo = *(const v8h*)p;        // K = kbase .. kbase+7
    v8h hi = *(const v8h*)(p + 16); // K = kbase+16 .. kbase+23
    return __builtin_shufflevector(lo, hi, 0, 1, 2, 3, 4, 5, 6, 7,
                                   8, 9, 10, 11, 12, 13, 14, 15);
}

// ---------------------------------------------------------------------------
// Kernel 0: repack conv weights (O,I,3,3) fp32 -> fp16, chunked B layout:
// Bw[kc][o][kk] with k = kc*32+kk = (ky*3+kx)*256 + c.  Each lane's B
// fragment is then two contiguous 16B runs in LDS.
// ---------------------------------------------------------------------------
__global__ void k_prep_w(const float* __restrict__ Wf, _Float16* __restrict__ Bw) {
    int e = blockIdx.x * 256 + threadIdx.x;
    if (e >= KTOT * CCH) return;
    int kc = e >> 13;           // chunk (8192 elements each)
    int r  = e & 8191;
    int o  = r >> 5;
    int kk = r & 31;
    int k  = kc * 32 + kk;
    int tap = k >> 8;
    int c   = k & 255;
    int ky = tap / 3, kx = tap % 3;
    Bw[e] = (_Float16)Wf[((o * CCH + c) * 3 + ky) * 3 + kx];
}

// ---------------------------------------------------------------------------
// Kernel 1: fused 3x3 conv (implicit GEMM via WMMA) + ReLU + score/delta
// heads + anchor decode + clip + min-size mask + sigmoid.
// One block = 16 pixels of one (image, level). 256 threads = 8 wave32,
// each wave owns 2 N-tiles (32 output channels).
// ---------------------------------------------------------------------------
__global__ __launch_bounds__(256) void k_conv_head(
    const float* __restrict__ f0, const float* __restrict__ f1,
    const float* __restrict__ f2, const float* __restrict__ f3,
    const float* __restrict__ f4, const float* __restrict__ im_shape,
    const _Float16* __restrict__ Bw, const float* __restrict__ b_feat,
    const float* __restrict__ W_score, const float* __restrict__ b_score,
    const float* __restrict__ W_delta, const float* __restrict__ b_delta,
    float* __restrict__ prob, float* __restrict__ boxes)
{
    __shared__ alignas(16) _Float16 sA[16 * 32];   // [m][kk]
    __shared__ alignas(16) _Float16 sB[256 * 32];  // [o][kk]
    __shared__ float sRF[16 * 256];
    __shared__ float sSD[16 * 16];

    const int tid  = threadIdx.x;
    const int lane = tid & 31;
    const int wave = tid >> 5;

    const int bi = blockIdx.x / TILES_PER_IMG;
    const int t  = blockIdx.x % TILES_PER_IMG;
    int lvl, tbase;
    if (t < 1600)      { lvl = 0; tbase = 0; }
    else if (t < 2000) { lvl = 1; tbase = 1600; }
    else if (t < 2100) { lvl = 2; tbase = 2000; }
    else if (t < 2125) { lvl = 3; tbase = 2100; }
    else               { lvl = 4; tbase = 2125; }
    const int H  = cHS[lvl];
    const int HW = H * H;
    const int p0 = (t - tbase) * 16;
    const float* feat;
    switch (lvl) { case 0: feat = f0; break; case 1: feat = f1; break;
                   case 2: feat = f2; break; case 3: feat = f3; break;
                   default: feat = f4; }
    const float* fb = feat + (size_t)bi * CCH * HW;

    // Per-thread A-gather coordinates, hoisted out of the K loop.
    // Thread loads elements (m0, kk) and (m0+8, kk).
    const int kkA = tid & 31;
    const int m0  = tid >> 5;
    const int m1  = m0 + 8;
    const int pA0 = p0 + m0, pA1 = p0 + m1;
    const bool v0 = pA0 < HW, v1 = pA1 < HW;
    const int hh0 = v0 ? (pA0 / H) : 0, ww0 = v0 ? (pA0 % H) : 0;
    const int hh1 = v1 ? (pA1 / H) : 0, ww1 = v1 ? (pA1 % H) : 0;

    v8f acc0 = {}; v8f acc1 = {};
    const int mrow  = lane & 15;
    const int kbase = ((lane >> 4) & 1) * 8;
    const int n0 = (wave * 2 + 0) * 16 + (lane & 15);
    const int n1 = (wave * 2 + 1) * 16 + (lane & 15);
    const _Float16* pAf  = sA + mrow * 32 + kbase;
    const _Float16* pBf0 = sB + n0 * 32 + kbase;
    const _Float16* pBf1 = sB + n1 * 32 + kbase;

    for (int kc = 0; kc < NCHUNK; ++kc) {
        const int k0  = kc * 32;
        const int tap = k0 >> 8;
        const int c0  = k0 & 255;
        const int ky  = tap / 3 - 1;
        const int kx  = tap % 3 - 1;
        // A tile: 16 pixels x 32 channels of this tap (SAME padding)
        {
            const int cch = c0 + kkA;
            int y0 = hh0 + ky, x0 = ww0 + kx;
            int y1 = hh1 + ky, x1 = ww1 + kx;
            float a0 = 0.f, a1 = 0.f;
            if (v0 && y0 >= 0 && y0 < H && x0 >= 0 && x0 < H)
                a0 = fb[(size_t)cch * HW + y0 * H + x0];
            if (v1 && y1 >= 0 && y1 < H && x1 >= 0 && x1 < H)
                a1 = fb[(size_t)cch * HW + y1 * H + x1];
            sA[m0 * 32 + kkA] = (_Float16)a0;
            sA[m1 * 32 + kkA] = (_Float16)a1;
        }
        // B tile: 256x32 fp16 chunk, fully contiguous -> 16B vector copies
        {
            const uint4* src = reinterpret_cast<const uint4*>(Bw + (size_t)kc * 8192);
            uint4* dst = reinterpret_cast<uint4*>(sB);
#pragma unroll
            for (int v4 = 0; v4 < 4; ++v4) dst[tid + v4 * 256] = src[tid + v4 * 256];
        }
        if (kc + 1 < NCHUNK)
            __builtin_prefetch(Bw + (size_t)(kc + 1) * 8192, 0, 1);
        __syncthreads();
        v16h af  = frag16(pAf);
        v16h bf0 = frag16(pBf0);
        v16h bf1 = frag16(pBf1);
        acc0 = __builtin_amdgcn_wmma_f32_16x16x32_f16(false, af, false, bf0,
                                                      (short)0, acc0, false, false);
        acc1 = __builtin_amdgcn_wmma_f32_16x16x32_f16(false, af, false, bf1,
                                                      (short)0, acc1, false, false);
        __syncthreads();
    }

    // Epilogue 1: bias + ReLU into LDS (C/D layout: lane = N + row-half)
    {
        const int mofs = ((lane >> 4) & 1) * 8;
#pragma unroll
        for (int r = 0; r < 8; ++r) {
            int M = r + mofs;
            float w0 = acc0[r] + b_feat[n0];
            float w1 = acc1[r] + b_feat[n1];
            sRF[M * 256 + n0] = w0 > 0.f ? w0 : 0.f;
            sRF[M * 256 + n1] = w1 > 0.f ? w1 : 0.f;
        }
    }
    __syncthreads();

    // Epilogue 2: 15 head outputs per pixel (3 scores + 12 deltas)
    if (tid < 240) {
        int px = tid / 15, o = tid % 15;
        const float* Wr = (o < 3) ? (W_score + o * CCH) : (W_delta + (o - 3) * CCH);
        float s = (o < 3) ? b_score[o] : b_delta[o - 3];
        const float* rf = sRF + px * 256;
#pragma unroll 8
        for (int c = 0; c < 256; ++c) s += rf[c] * Wr[c];
        sSD[px * 16 + o] = s;
    }
    __syncthreads();

    // Epilogue 3: decode anchors, clip, min-size mask, sigmoid, store
    if (tid < 48) {
        int px = tid / 3, a = tid % 3;
        int p = p0 + px;
        if (p < HW) {
            int hh = p / H, ww = p % H;
            const float sr[3] = {0.70710678118654752f, 1.0f, 1.41421356237309505f};
            float stride = cSTRIDE[lvl], size = cSIZE[lvl];
            float aw = size / sr[a], ah = size * sr[a];
            float acx = (ww + 0.5f) * stride, acy = (hh + 0.5f) * stride;
            float sc = sSD[px * 16 + a];
            float dx = sSD[px * 16 + 3 + 4 * a + 0];
            float dy = sSD[px * 16 + 3 + 4 * a + 1];
            float dw = fminf(sSD[px * 16 + 3 + 4 * a + 2], BBOX_CLIP_F);
            float dh = fminf(sSD[px * 16 + 3 + 4 * a + 3], BBOX_CLIP_F);
            float cx = dx * aw + acx, cy = dy * ah + acy;
            float w2 = expf(dw) * aw, h2 = expf(dh) * ah;
            float x1 = cx - 0.5f * w2, y1 = cy - 0.5f * h2;
            float x2 = cx + 0.5f * w2, y2 = cy + 0.5f * h2;
            float imh = im_shape[bi * 2 + 0], imw = im_shape[bi * 2 + 1];
            x1 = fminf(fmaxf(x1, 0.f), imw); x2 = fminf(fmaxf(x2, 0.f), imw);
            y1 = fminf(fmaxf(y1, 0.f), imh); y2 = fminf(fmaxf(y2, 0.f), imh);
            bool ok = (x2 - x1 >= MIN_SZ) && (y2 - y1 >= MIN_SZ);
            float pr = ok ? (1.f / (1.f + expf(-sc))) : NEGV;
            size_t gi = (size_t)bi * TOTANCH + cLOFF[lvl] + (size_t)p * 3 + a;
            prob[gi] = pr;
            float* bo = boxes + gi * 4;
            bo[0] = x1; bo[1] = y1; bo[2] = x2; bo[3] = y2;
        }
    }
}

// ---------------------------------------------------------------------------
// Kernel 2: per (image, level) — deterministic top-k1 (radix threshold search
// + ordered compaction, ties by lowest index), bitonic sort, greedy NMS,
// emit top-300 kept entries padded with NEG.
// ---------------------------------------------------------------------------
__global__ __launch_bounds__(1024) void k_nms_level(
    const float* __restrict__ prob, const float* __restrict__ boxes,
    float* __restrict__ lvlout)
{
    __shared__ unsigned int sKey[1024];
    __shared__ int sIdx[1024];
    __shared__ float sBox[1024 * 4];
    __shared__ float sPr[1024];
    __shared__ float sArea[1024];
    __shared__ unsigned char sSupp[1024];
    __shared__ int sCnt[2];
    __shared__ int sWarp[32];

    const int tid  = threadIdx.x;
    const int lane = tid & 31;
    const int wid  = tid >> 5;
    const int b    = blockIdx.x / NLVL;
    const int lvl  = blockIdx.x % NLVL;
    const int H    = cHS[lvl];
    const int N    = H * H * 3;
    const int k1   = N < PRE_NMS_K ? N : PRE_NMS_K;
    const float* pr = prob + (size_t)b * TOTANCH + cLOFF[lvl];
    const float* bx = boxes + ((size_t)b * TOTANCH + cLOFF[lvl]) * 4;

    auto keyOf = [&](int i) -> unsigned {
        unsigned u = __float_as_uint(pr[i]);
        return (u & 0x80000000u) ? ~u : (u | 0x80000000u);
    };

    // --- radix threshold search: Tpref = k1-th largest key
    unsigned Tpref = 0u;
    for (int bit = 31; bit >= 0; --bit) {
        unsigned cand = Tpref | (1u << bit);
        if (tid == 0) sCnt[0] = 0;
        __syncthreads();
        int c = 0;
        for (int i = tid; i < N; i += 1024) c += (keyOf(i) >= cand) ? 1 : 0;
        if (c) atomicAdd(&sCnt[0], c);
        __syncthreads();
        if (sCnt[0] >= k1) Tpref = cand;
        __syncthreads();
    }
    if (tid == 0) sCnt[1] = 0;
    __syncthreads();
    {
        int c = 0;
        for (int i = tid; i < N; i += 1024) c += (keyOf(i) > Tpref) ? 1 : 0;
        if (c) atomicAdd(&sCnt[1], c);
    }
    __syncthreads();
    const int cntGT  = sCnt[1];
    const int needEQ = k1 - cntGT;
    __syncthreads();

    // --- ordered compaction (GT all, EQ first needEQ in index order)
    if (tid == 0) { sCnt[0] = 0; sCnt[1] = 0; }
    __syncthreads();
    for (int base = 0; base < N; base += 1024) {
        int i = base + tid;
        bool inr = (i < N);
        unsigned u = inr ? keyOf(i) : 0u;
        bool gt = inr && (u > Tpref);
        bool eq = inr && (u == Tpref);
        unsigned long long bg = __ballot(gt);
        int lp = __popcll(bg & ((1ull << lane) - 1ull));
        if (lane == 0) sWarp[wid] = __popcll(bg);
        __syncthreads();
        int wp = 0, chunkGT = 0;
        for (int w = 0; w < 32; ++w) { int v = sWarp[w]; if (w < wid) wp += v; chunkGT += v; }
        int gtB = sCnt[0];
        if (gt) { int pos = gtB + wp + lp; sKey[pos] = u; sIdx[pos] = i; }
        __syncthreads();
        unsigned long long be = __ballot(eq);
        int lpe = __popcll(be & ((1ull << lane) - 1ull));
        if (lane == 0) sWarp[wid] = __popcll(be);
        __syncthreads();
        int wpe = 0, chunkEQ = 0;
        for (int w = 0; w < 32; ++w) { int v = sWarp[w]; if (w < wid) wpe += v; chunkEQ += v; }
        int eqB = sCnt[1];
        if (eq) {
            int r = eqB + wpe + lpe;
            if (r < needEQ) { int pos = cntGT + r; sKey[pos] = u; sIdx[pos] = i; }
        }
        __syncthreads();
        if (tid == 0) { sCnt[0] = gtB + chunkGT; sCnt[1] = eqB + chunkEQ; }
        __syncthreads();
    }
    if (tid >= k1) { sKey[tid] = 0u; sIdx[tid] = 0x7FFFFFFF; }
    __syncthreads();

    // --- bitonic sort 1024: key desc, idx asc
    for (int size = 2; size <= 1024; size <<= 1) {
        for (int stride = size >> 1; stride > 0; stride >>= 1) {
            int i = tid, j = i ^ stride;
            if (j > i) {
                bool desc = ((i & size) == 0);
                unsigned ka = sKey[i], kb = sKey[j];
                int ia = sIdx[i], ib = sIdx[j];
                bool aFirst = (ka > kb) || (ka == kb && ia < ib);
                if (desc ? !aFirst : aFirst) {
                    sKey[i] = kb; sKey[j] = ka; sIdx[i] = ib; sIdx[j] = ia;
                }
            }
            __syncthreads();
        }
    }

    // --- gather boxes + probs
    if (tid < k1) {
        int gi = sIdx[tid];
        const float* bb = bx + (size_t)gi * 4;
        float x1 = bb[0], y1 = bb[1], x2 = bb[2], y2 = bb[3];
        sBox[tid * 4 + 0] = x1; sBox[tid * 4 + 1] = y1;
        sBox[tid * 4 + 2] = x2; sBox[tid * 4 + 3] = y2;
        sArea[tid] = (x2 - x1) * (y2 - y1);
        unsigned u = sKey[tid];
        sPr[tid] = (u & 0x80000000u) ? __uint_as_float(u & 0x7FFFFFFFu)
                                     : __uint_as_float(~u);
    } else {
        sPr[tid] = NEGV; sArea[tid] = 0.f;
        sBox[tid * 4 + 0] = 0.f; sBox[tid * 4 + 1] = 0.f;
        sBox[tid * 4 + 2] = 0.f; sBox[tid * 4 + 3] = 0.f;
    }
    sSupp[tid] = 0;
    __syncthreads();

    // --- greedy NMS (matches reference suppression loop)
    for (int i = 0; i < k1; ++i) {
        bool act = (sSupp[i] == 0) && (sPr[i] > 0.f);
        if (act) {
            int j = tid;
            if (j > i && j < k1) {
                float lx = fmaxf(sBox[i * 4 + 0], sBox[j * 4 + 0]);
                float ly = fmaxf(sBox[i * 4 + 1], sBox[j * 4 + 1]);
                float rx = fminf(sBox[i * 4 + 2], sBox[j * 4 + 2]);
                float ry = fminf(sBox[i * 4 + 3], sBox[j * 4 + 3]);
                float iw = fmaxf(rx - lx, 0.f), ih = fmaxf(ry - ly, 0.f);
                float inter = iw * ih;
                float iou = inter / (sArea[i] + sArea[j] - inter + 1e-9f);
                if (iou > NMS_TH) sSupp[j] = 1;
            }
        }
        __syncthreads();
    }

    // --- emit kept entries (already score-descending), pad with NEG
    bool kept = (tid < k1) && (sSupp[tid] == 0) && (sPr[tid] > 0.f);
    unsigned long long bk = __ballot(kept);
    int lpk = __popcll(bk & ((1ull << lane) - 1ull));
    if (lane == 0) sWarp[wid] = __popcll(bk);
    __syncthreads();
    int wpk = 0, total = 0;
    for (int w = 0; w < 32; ++w) { int v = sWarp[w]; if (w < wid) wpk += v; total += v; }
    int rank = wpk + lpk;
    float* LO = lvlout + (size_t)(b * NLVL + lvl) * POST_NMS_K * 5;
    if (kept && rank < POST_NMS_K) {
        LO[rank * 5 + 0] = sBox[tid * 4 + 0];
        LO[rank * 5 + 1] = sBox[tid * 4 + 1];
        LO[rank * 5 + 2] = sBox[tid * 4 + 2];
        LO[rank * 5 + 3] = sBox[tid * 4 + 3];
        LO[rank * 5 + 4] = sPr[tid];
    }
    int keptC = total < POST_NMS_K ? total : POST_NMS_K;
    if (tid < POST_NMS_K && tid >= keptC) {
        LO[tid * 5 + 0] = 0.f; LO[tid * 5 + 1] = 0.f;
        LO[tid * 5 + 2] = 0.f; LO[tid * 5 + 3] = 0.f;
        LO[tid * 5 + 4] = NEGV;
    }
}

// ---------------------------------------------------------------------------
// Kernel 3: per image — merge 5*300 entries, top-300 overall, zero invalid,
// count valid.
// ---------------------------------------------------------------------------
__global__ __launch_bounds__(1024) void k_final(
    const float* __restrict__ lvlout, float* __restrict__ outp,
    int* __restrict__ outn)
{
    __shared__ float k3[2048];
    __shared__ int i3[2048];
    __shared__ int sCnt;
    const int tid = threadIdx.x;
    const int b = blockIdx.x;
    const float* LI = lvlout + (size_t)b * (NLVL * POST_NMS_K) * 5;
    for (int t = tid; t < 2048; t += 1024) {
        if (t < NLVL * POST_NMS_K) { k3[t] = LI[t * 5 + 4]; i3[t] = t; }
        else                       { k3[t] = -3.0f; i3[t] = 0x7FFFFFFF; }
    }
    __syncthreads();
    for (int size = 2; size <= 2048; size <<= 1) {
        for (int stride = size >> 1; stride > 0; stride >>= 1) {
            for (int t = tid; t < 2048; t += 1024) {
                int j = t ^ stride;
                if (j > t) {
                    bool desc = ((t & size) == 0);
                    float ka = k3[t], kb = k3[j];
                    int ia = i3[t], ib = i3[j];
                    bool aFirst = (ka > kb) || (ka == kb && ia < ib);
                    if (desc ? !aFirst : aFirst) {
                        k3[t] = kb; k3[j] = ka; i3[t] = ib; i3[j] = ia;
                    }
                }
            }
            __syncthreads();
        }
    }
    if (tid == 0) sCnt = 0;
    __syncthreads();
    if (tid < POST_NMS_K) {
        float p = k3[tid];
        int src = i3[tid];
        bool v = p > 0.f;
        float* dst = outp + (size_t)(b * POST_NMS_K + tid) * 5;
        if (v) {
            dst[0] = LI[src * 5 + 0]; dst[1] = LI[src * 5 + 1];
            dst[2] = LI[src * 5 + 2]; dst[3] = LI[src * 5 + 3];
            dst[4] = p;
            atomicAdd(&sCnt, 1);
        } else {
            dst[0] = 0.f; dst[1] = 0.f; dst[2] = 0.f; dst[3] = 0.f; dst[4] = 0.f;
        }
    }
    __syncthreads();
    if (tid == 0) outn[b] = sCnt;
}

// ---------------------------------------------------------------------------
extern "C" void kernel_launch(void* const* d_in, const int* in_sizes, int n_in,
                              void* d_out, int out_size, void* d_ws, size_t ws_size,
                              hipStream_t stream)
{
    const float* f0       = (const float*)d_in[0];
    const float* f1       = (const float*)d_in[1];
    const float* f2       = (const float*)d_in[2];
    const float* f3       = (const float*)d_in[3];
    const float* f4       = (const float*)d_in[4];
    const float* im_shape = (const float*)d_in[5];
    const float* W_feat   = (const float*)d_in[6];
    const float* b_feat   = (const float*)d_in[7];
    const float* W_score  = (const float*)d_in[8];
    const float* b_score  = (const float*)d_in[9];
    const float* W_delta  = (const float*)d_in[10];
    const float* b_delta  = (const float*)d_in[11];

    const int Bn = in_sizes[5] / 2;  // B from im_shape (B,2)

    // workspace layout (all 16B aligned)
    char* ws = (char*)d_ws;
    _Float16* Bw   = (_Float16*)ws;                                   // 1,179,648 B
    float* prob    = (float*)(ws + 1179648);                          //   818,400 B (B=2)
    float* boxes   = (float*)(ws + 1179648 + 818400);                 // 3,273,600 B
    float* lvlout  = (float*)(ws + 1179648 + 818400 + 3273600);       //    60,000 B
    (void)ws_size; (void)n_in; (void)out_size;

    k_prep_w<<<(KTOT * CCH) / 256, 256, 0, stream>>>(W_feat, Bw);
    k_conv_head<<<Bn * TILES_PER_IMG, 256, 0, stream>>>(
        f0, f1, f2, f3, f4, im_shape, Bw, b_feat, W_score, b_score,
        W_delta, b_delta, prob, boxes);
    k_nms_level<<<Bn * NLVL, 1024, 0, stream>>>(prob, boxes, lvlout);

    float* outp = (float*)d_out;
    int* outn = (int*)(outp + (size_t)Bn * POST_NMS_K * 5);
    k_final<<<Bn, 1024, 0, stream>>>(lvlout, outp, outn);
}
